// SelfAttention_83537113907945
// MI455X (gfx1250) — matrix operand order repaired and checked by reading, not compiled, not run on hardware
//
#include <hip/hip_runtime.h>

#define B_ 8
#define E_ 128
#define S_ 4096

typedef __attribute__((ext_vector_type(16))) _Float16 v16h;
typedef __attribute__((ext_vector_type(8)))  _Float16 v8h;
typedef __attribute__((ext_vector_type(8)))  float    v8f;
typedef __attribute__((ext_vector_type(4)))  int      v4i;

// ---------------------------------------------------------------------------
// WMMA wrapper: D = A(16x32 f16) * B(32x16 f16) + C(16x16 f32)
// ---------------------------------------------------------------------------
__device__ __forceinline__ v8f wmma_f16(v16h a, v16h b, v8f c) {
    return __builtin_amdgcn_wmma_f32_16x16x32_f16(
        /*neg_a=*/false, a, /*neg_b=*/false, b,
        /*c_mod=*/(short)0, c, /*reuse_a=*/false, /*reuse_b=*/false);
}

// ---------------------------------------------------------------------------
// A-fragment (16x32, 16-bit) from row-major storage[m][k] with row stride ld.
// lane<16 -> M=lane, halfs 0-7 = K 0..7, halfs 8-15 = K 16..23
// lane>=16 -> M=lane-16, K shifted by +8.  Two 16B loads per lane.
// ---------------------------------------------------------------------------
__device__ __forceinline__ v16h load_frag_a(const _Float16* p, int ld) {
    const int lane = threadIdx.x & 31;
    const int row  = lane & 15;
    const int cb   = (lane >> 4) << 3;      // 0 or 8
    const _Float16* r = p + row * ld + cb;
    v8h lo = *(const v8h*)(r);
    v8h hi = *(const v8h*)(r + 16);
    v16h out;
#pragma unroll
    for (int i = 0; i < 8; ++i) { out[i] = lo[i]; out[i + 8] = hi[i]; }
    return out;
}

// ---------------------------------------------------------------------------
// B-fragment (32x16, 16-bit) from N-major storage[n][k] with row stride ld.
// lane<16 -> N=lane, halfs 0-7 = K 0..7, halfs 8-15 = K 16..23
// lane>=16 -> N=lane-16, K shifted by +8.  Two 16B loads per lane.
// ---------------------------------------------------------------------------
__device__ __forceinline__ v16h load_frag_b(const _Float16* p, int ld) {
    const int lane = threadIdx.x & 31;
    const int row  = lane & 15;
    const int cb   = (lane >> 4) << 3;      // 0 or 8
    const _Float16* r = p + row * ld + cb;
    v8h lo = *(const v8h*)(r);
    v8h hi = *(const v8h*)(r + 16);
    v16h out;
#pragma unroll
    for (int i = 0; i < 8; ++i) { out[i] = lo[i]; out[i + 8] = hi[i]; }
    return out;
}

// ---------------------------------------------------------------------------
// Async 16-byte global -> LDS copy (ASYNCcnt-tracked, gfx1250)
// ---------------------------------------------------------------------------
__device__ __forceinline__ void async_copy16(const _Float16* g, _Float16* l) {
    __builtin_amdgcn_global_load_async_to_lds_b128((v4i*)g, (v4i*)l, 0, 0);
}

// ---------------------------------------------------------------------------
// Kernel 1: x[B,E,S] f32 -> xT[B,S,E] f16 (LDS tile transpose, coalesced)
// ---------------------------------------------------------------------------
__global__ void __launch_bounds__(256)
transpose_cvt_kernel(const float* __restrict__ x, _Float16* __restrict__ xT) {
    __shared__ float tile[32][33];
    const int b  = blockIdx.z;
    const int e0 = blockIdx.y * 32;
    const int s0 = blockIdx.x * 32;
    const int tx = threadIdx.x, ty = threadIdx.y;
#pragma unroll
    for (int i = 0; i < 4; ++i)
        tile[ty + 8 * i][tx] =
            x[((size_t)(b * E_ + e0 + ty + 8 * i)) * S_ + s0 + tx];
    __syncthreads();
#pragma unroll
    for (int i = 0; i < 4; ++i)
        xT[((size_t)(b * S_ + s0 + ty + 8 * i)) * E_ + e0 + tx] =
            (_Float16)tile[tx][ty + 8 * i];
}

// ---------------------------------------------------------------------------
// Kernel 2: elementwise f32 -> f16 (for weight matrices)
// ---------------------------------------------------------------------------
__global__ void cvt_f16_kernel(const float* __restrict__ src,
                               _Float16* __restrict__ dst, int n) {
    int i = blockIdx.x * 256 + threadIdx.x;
    if (i < n) dst[i] = (_Float16)src[i];
}

// ---------------------------------------------------------------------------
// Kernel 3: QKV projection.  One wave = one 16(s) x 16(f) tile of Q,K,V.
// ---------------------------------------------------------------------------
__global__ void __launch_bounds__(128)
qkv_project_kernel(const _Float16* __restrict__ xT,
                   const _Float16* __restrict__ Wqh,
                   const _Float16* __restrict__ Wkh,
                   const _Float16* __restrict__ Wvh,
                   const float* __restrict__ bq,
                   const float* __restrict__ bk,
                   const float* __restrict__ bv,
                   _Float16* __restrict__ Qh,
                   _Float16* __restrict__ Kh,
                   _Float16* __restrict__ Vt) {
    const int lane = threadIdx.x & 31;
    const int wave = threadIdx.x >> 5;
    const int gid  = blockIdx.x * 4 + wave;       // wave id: B * 256 * 8
    const int b    = gid >> 11;
    const int rem  = gid & 2047;
    const int st   = rem >> 3;                    // s-tile 0..255
    const int ft   = rem & 7;                     // f-tile 0..7

    const _Float16* xTb = xT + (size_t)b * S_ * E_ + (size_t)st * 16 * E_;
    v8f aq = {}, ak = {}, av = {};
#pragma unroll
    for (int kk = 0; kk < 4; ++kk) {
        v16h a  = load_frag_a(xTb + kk * 32, E_);
        v16h wq = load_frag_b(Wqh + ft * 16 * E_ + kk * 32, E_);
        v16h wk = load_frag_b(Wkh + ft * 16 * E_ + kk * 32, E_);
        v16h wv = load_frag_b(Wvh + ft * 16 * E_ + kk * 32, E_);
        aq = wmma_f16(a, wq, aq);
        ak = wmma_f16(a, wk, ak);
        av = wmma_f16(a, wv, av);
    }
    const int fcol = ft * 16 + (lane & 15);
    const float bqv = bq[fcol], bkv = bk[fcol], bvv = bv[fcol];
    _Float16* Qb = Qh + (size_t)b * S_ * E_;
    _Float16* Kb = Kh + (size_t)b * S_ * E_;
    _Float16* Vb = Vt + (size_t)b * E_ * S_;
    const int rbase = st * 16 + ((lane >> 4) << 3);   // C-layout: M = r + 8*(lane/16)
#pragma unroll
    for (int r = 0; r < 8; ++r) {
        const int row = rbase + r;
        Qb[(size_t)row * E_ + fcol] = (_Float16)(aq[r] + bqv);
        Kb[(size_t)row * E_ + fcol] = (_Float16)(ak[r] + bkv);
        Vb[(size_t)fcol * S_ + row] = (_Float16)(av[r] + bvv);
    }
}

// ---------------------------------------------------------------------------
// Kernel 4: flash attention with async double-buffered K/V staging in LDS.
// Block = 4 waves, one batch, 64 consecutive query rows (16/wave).
// Per 32-key step: K tile 32x128 + V tile 128x32 staged once per block.
// ---------------------------------------------------------------------------
#define LDK 136   // K tile row stride (halfs): 272B = 17*16B, banks 4*row%64 distinct
#define LDV 40    // V/P row stride (halfs):     80B =  5*16B, banks 20*row%64 distinct

__global__ void __launch_bounds__(128)
flash_attn_kernel(const _Float16* __restrict__ Qh,
                  const _Float16* __restrict__ Kh,
                  const _Float16* __restrict__ Vt,
                  float* __restrict__ out) {
    __shared__ __align__(16) _Float16 Kbuf[2][32 * LDK];
    __shared__ __align__(16) _Float16 Vbuf[2][128 * LDV];
    __shared__ __align__(16) _Float16 Pshm[4][16 * LDV];

    const int tid  = threadIdx.x;
    const int lane = tid & 31;
    const int wave = tid >> 5;
    const int b    = blockIdx.x >> 6;               // 64 blocks per batch
    const int qt   = (blockIdx.x & 63) * 4 + wave;  // q-tile of 16 rows

    const _Float16* Qb = Qh + (size_t)b * S_ * E_ + (size_t)qt * 16 * E_;
    const _Float16* Kb = Kh + (size_t)b * S_ * E_;
    const _Float16* Vb = Vt + (size_t)b * E_ * S_;
    _Float16* P = &Pshm[wave][0];

    // ---- issue async prefetch of tile j into LDS buffer buf (8 loads/thread)
    auto prefetch = [&](int j, int buf) {
        _Float16* Kl = &Kbuf[buf][0];
        _Float16* Vl = &Vbuf[buf][0];
#pragma unroll
        for (int i = 0; i < 4; ++i) {               // K: 32 rows x 128 halfs
            const int c  = i * 128 + tid;           // 512 16B chunks
            const int kr = c >> 4, kc = (c & 15) << 3;
            async_copy16(Kb + (size_t)(j * 32 + kr) * E_ + kc, Kl + kr * LDK + kc);
        }
#pragma unroll
        for (int i = 0; i < 4; ++i) {               // V: 128 rows x 32 halfs
            const int c  = i * 128 + tid;
            const int vr = c >> 2, vc = (c & 3) << 3;
            async_copy16(Vb + (size_t)vr * S_ + j * 32 + vc, Vl + vr * LDV + vc);
        }
    };

    prefetch(0, 0);

    v16h qf[4];
#pragma unroll
    for (int kk = 0; kk < 4; ++kk) qf[kk] = load_frag_a(Qb + kk * 32, E_);

    v8f o[8];
#pragma unroll
    for (int i = 0; i < 8; ++i) o[i] = (v8f){};
    float m[8], l[8];
#pragma unroll
    for (int r = 0; r < 8; ++r) { m[r] = -1e30f; l[r] = 0.0f; }

    const float C = 1.44269504088896f * 0.08838834764831845f; // log2(e)/sqrt(128)
    const int  pcol  = lane & 15;
    const int  rhigh = (lane >> 4) << 3;            // C-layout: M = r + rhigh

    constexpr int NJ = S_ / 32;
    for (int j = 0; j < NJ; ++j) {
        if (j + 1 < NJ) {
            prefetch(j + 1, (j + 1) & 1);
            asm volatile("s_wait_asynccnt 0x8" ::: "memory");  // tile j complete
        } else {
            asm volatile("s_wait_asynccnt 0x0" ::: "memory");
        }
        __syncthreads();                             // tile j visible to all waves

        const _Float16* Kl = &Kbuf[j & 1][0];
        const _Float16* Vl = &Vbuf[j & 1][0];

        // ---- scores: two 16x16 tiles over K=E=128 ----
        v8f s0 = (v8f){}, s1 = (v8f){};
#pragma unroll
        for (int kk = 0; kk < 4; ++kk) {
            v16h k0 = load_frag_b(Kl + kk * 32, LDK);
            v16h k1 = load_frag_b(Kl + 16 * LDK + kk * 32, LDK);
            s0 = wmma_f16(qf[kk], k0, s0);
            s1 = wmma_f16(qf[kk], k1, s1);
        }
        // ---- online softmax update (base-2 domain) ----
#pragma unroll
        for (int r = 0; r < 8; ++r) {
            float z0 = s0[r] * C, z1 = s1[r] * C;
            float v = fmaxf(z0, z1);
            v = fmaxf(v, __shfl_xor(v, 1));
            v = fmaxf(v, __shfl_xor(v, 2));
            v = fmaxf(v, __shfl_xor(v, 4));
            v = fmaxf(v, __shfl_xor(v, 8));
            float mn   = fmaxf(m[r], v);
            float corr = __builtin_exp2f(m[r] - mn);
            float p0   = __builtin_exp2f(z0 - mn);
            float p1   = __builtin_exp2f(z1 - mn);
            float rs = p0 + p1;
            rs += __shfl_xor(rs, 1);
            rs += __shfl_xor(rs, 2);
            rs += __shfl_xor(rs, 4);
            rs += __shfl_xor(rs, 8);
            l[r] = l[r] * corr + rs;
            m[r] = mn;
#pragma unroll
            for (int t = 0; t < 8; ++t) o[t][r] *= corr;
            const int prow = rhigh + r;
            P[prow * LDV + pcol]      = (_Float16)p0;
            P[prow * LDV + 16 + pcol] = (_Float16)p1;
        }
        asm volatile("s_wait_dscnt 0" ::: "memory");
        // ---- O += P (16x32) * V (32x128) ----
        v16h pf = load_frag_a(P, LDV);
#pragma unroll
        for (int et = 0; et < 8; ++et) {
            v16h vf = load_frag_b(Vl + (et * 16) * LDV, LDV);
            o[et] = wmma_f16(pf, vf, o[et]);
        }
        __syncthreads();                             // done reading tile j's buffer
    }

    // ---- finalize: divide by row sums, store out[b, e, q] (fp32) ----
    float* ob = out + (size_t)b * E_ * S_;
    const int q0 = qt * 16 + rhigh;
#pragma unroll
    for (int r = 0; r < 8; ++r) {
        const float inv = 1.0f / l[r];
#pragma unroll
        for (int et = 0; et < 8; ++et) {
            const int e = et * 16 + pcol;
            ob[(size_t)e * S_ + q0 + r] = o[et][r] * inv;
        }
    }
}

// ---------------------------------------------------------------------------
extern "C" void kernel_launch(void* const* d_in, const int* in_sizes, int n_in,
                              void* d_out, int out_size, void* d_ws, size_t ws_size,
                              hipStream_t stream) {
    const float* x  = (const float*)d_in[0];
    const float* Wq = (const float*)d_in[1];
    const float* bq = (const float*)d_in[2];
    const float* Wk = (const float*)d_in[3];
    const float* bk = (const float*)d_in[4];
    const float* Wv = (const float*)d_in[5];
    const float* bv = (const float*)d_in[6];
    float* out = (float*)d_out;

    const size_t tensor_bytes = (size_t)B_ * S_ * E_ * sizeof(_Float16); // 8.39 MB
    const size_t w_bytes      = (size_t)E_ * E_ * sizeof(_Float16);
    char* ws = (char*)d_ws;
    _Float16* xT  = (_Float16*)(ws);
    _Float16* Qh  = (_Float16*)(ws + 1 * tensor_bytes);
    _Float16* Kh  = (_Float16*)(ws + 2 * tensor_bytes);
    _Float16* Vt  = (_Float16*)(ws + 3 * tensor_bytes);
    _Float16* Wqh = (_Float16*)(ws + 4 * tensor_bytes);
    _Float16* Wkh = (_Float16*)(ws + 4 * tensor_bytes + 1 * w_bytes);
    _Float16* Wvh = (_Float16*)(ws + 4 * tensor_bytes + 2 * w_bytes);

    transpose_cvt_kernel<<<dim3(S_ / 32, E_ / 32, B_), dim3(32, 8), 0, stream>>>(x, xT);
    const int wn = E_ * E_;
    cvt_f16_kernel<<<(wn + 255) / 256, 256, 0, stream>>>(Wq, Wqh, wn);
    cvt_f16_kernel<<<(wn + 255) / 256, 256, 0, stream>>>(Wk, Wkh, wn);
    cvt_f16_kernel<<<(wn + 255) / 256, 256, 0, stream>>>(Wv, Wvh, wn);
    qkv_project_kernel<<<(B_ * (S_ / 16) * (E_ / 16)) / 4, 128, 0, stream>>>(
        xT, Wqh, Wkh, Wvh, bq, bk, bv, Qh, Kh, Vt);
    flash_attn_kernel<<<(B_ * (S_ / 16)) / 4, 128, 0, stream>>>(Qh, Kh, Vt, out);
}
